// vnDGCNN_2894807957694
// MI455X (gfx1250) — matrix operand order repaired and checked by reading, not compiled, not run on hardware
//
#include <hip/hip_runtime.h>
#include <hip/hip_bf16.h>
#include <math.h>

// ---------------- problem constants (from reference) ----------------
#define Bsz   2
#define Npts  4096
#define KNN   20
// C64 = 21, C1024 = 341, C1024/2 = 170, FFD=256, FFD+FREE=260, classes=19

typedef __attribute__((ext_vector_type(16))) __bf16 v16bf;
typedef __attribute__((ext_vector_type(8)))  __bf16 v8bf;
typedef __attribute__((ext_vector_type(8)))  float  v8f;

// ---------------- small helpers ----------------
__device__ __forceinline__ float act_mode(float x, int mode) {
    if (mode == 1) return x > 0.f ? x : 0.2f * x;              // leaky 0.2
    if (mode == 2) return x > 0.f ? x : 0.1f * x;              // leaky 0.1
    if (mode == 3) { float y = x > 0.f ? x : 0.1f * x;          // sigmoid(leaky 0.1)
                     return 1.f / (1.f + __expf(-y)); }
    if (mode == 4) { float y = x > 0.f ? x : 0.1f * x;          // relu(leaky 0.1)
                     return y > 0.f ? y : 0.f; }
    return x;
}

__global__ void k_zero(float* p, int n) {
    int i = blockIdx.x * blockDim.x + threadIdx.x;
    if (i < n) p[i] = 0.f;
}

// (B,N,3) -> (B,3,N)
__global__ void k_transpose_x(const float* __restrict__ x, float* __restrict__ xt) {
    int i = blockIdx.x * blockDim.x + threadIdx.x;
    int total = Bsz * Npts * 3;
    if (i >= total) return;
    int d = i % 3; int n = (i / 3) % Npts; int b = i / (3 * Npts);
    xt[((long)b * 3 + d) * Npts + n] = x[i];
}

// per-point squared norms over D feature rows (row stride Npts, batch stride bs)
__global__ void k_xx(const float* __restrict__ X, int D, long bs, float* __restrict__ xx) {
    int i = blockIdx.x * blockDim.x + threadIdx.x;
    if (i >= Bsz * Npts) return;
    int n = i % Npts, b = i / Npts;
    const float* p = X + (long)b * bs;
    float s = 0.f;
    for (int d = 0; d < D; ++d) { float v = p[(long)d * Npts + n]; s += v * v; }
    xx[i] = s;
}

// one workgroup per (b, i): build pd row in LDS, then 20 argmax passes
__global__ __launch_bounds__(256) void k_knn(const float* __restrict__ X,
                                             const float* __restrict__ xx,
                                             int D, long bs, int* __restrict__ idxout) {
    __shared__ float pd[Npts];
    __shared__ float xi[64];
    __shared__ float rv[256];
    __shared__ int   ri[256];
    int b = blockIdx.y, i = blockIdx.x, tid = threadIdx.x;
    const float* Xb = X + (long)b * bs;
    if (tid < D) xi[tid] = Xb[(long)tid * Npts + i];
    __syncthreads();
    float xxi = xx[b * Npts + i];
    for (int j = tid; j < Npts; j += 256) {
        float dot = 0.f;
        for (int d = 0; d < D; ++d) dot += xi[d] * Xb[(long)d * Npts + j];
        pd[j] = 2.f * dot - xxi - xx[b * Npts + j];
    }
    __syncthreads();
    for (int t = 0; t < KNN; ++t) {
        float best = -3.4e38f; int bj = 0;
        for (int j = tid; j < Npts; j += 256) {
            float v = pd[j];
            if (v > best) { best = v; bj = j; }
        }
        rv[tid] = best; ri[tid] = bj;
        __syncthreads();
        for (int off = 128; off > 0; off >>= 1) {
            if (tid < off) {
                if (rv[tid + off] > rv[tid] ||
                    (rv[tid + off] == rv[tid] && ri[tid + off] < ri[tid])) {
                    rv[tid] = rv[tid + off]; ri[tid] = ri[tid + off];
                }
            }
            __syncthreads();
        }
        if (tid == 0) {
            idxout[((long)(b * Npts) + i) * KNN + t] = ri[0];
            pd[ri[0]] = -3.4e38f;
        }
        __syncthreads();
    }
}

// F (B,C,3,N) view w/ batch stride bs -> out (B,2C,3,N,K)
__global__ void k_graphfeat(const float* __restrict__ F, int C, long bs,
                            const int* __restrict__ idx, float* __restrict__ out) {
    long i = blockIdx.x * (long)blockDim.x + threadIdx.x;
    long total = (long)Bsz * 2 * C * 3 * Npts * KNN;
    if (i >= total) return;
    int k = i % KNN; long r = i / KNN;
    int n = r % Npts; r /= Npts;
    int v = r % 3;    r /= 3;
    int c = r % (2 * C);
    int b = r / (2 * C);
    const float* Fb = F + (long)b * bs;
    float xn = Fb[((long)(c % C) * 3 + v) * Npts + n];
    float val;
    if (c < C) {
        int e = idx[((long)(b * Npts) + n) * KNN + k];
        val = Fb[((long)c * 3 + v) * Npts + e] - xn;
    } else val = xn;
    out[i] = val;
}

// ---------------- WMMA GEMM: C[b] = act(A(MxK) * B[b](KxP) + bias) ----------------
// Workgroup tile 32(M) x 128(P); 8 waves, each wave = 16x32 (two 16x16 accumulators
// sharing one A fragment). f32 -> bf16 staged in LDS in *fragment-swizzled* order so
// every lane's fragment is 2 contiguous 16B ds_load_b128s. Out-of-range staging is
// handled by clamped addresses * 0/1 mask (multiply, NOT select, so LLVM cannot sink
// the loads into divergent exec regions). Requires: P % 128 == 0 and B rows 16B-aligned.
__global__ __launch_bounds__(256) void k_gemm(const float* __restrict__ A,
                                              const float* __restrict__ Bm,
                                              float* __restrict__ Cm,
                                              const float* __restrict__ bias,
                                              int M, int Kc, int P,
                                              long strideB, long strideC,
                                              int mode, int transC) {
    // row stride 40 bf16 (80B, keeps 16B alignment of fragments, staggers banks)
    __shared__ __align__(16) __bf16 lAs[32 * 40];
    __shared__ __align__(16) __bf16 lBs[128 * 40];
    int tid   = threadIdx.x;
    int b     = blockIdx.z;
    int mBase = blockIdx.y * 32;
    int nBase = blockIdx.x * 128;
    const float* Bb = Bm + (long)b * strideB;
    int lane = tid & 31;
    int wave = tid >> 5;
    int m = lane & 15;
    int h = lane >> 4;
    int wm = (wave >> 2) * 16;   // 2 wave-rows of M
    int wn = (wave & 3) * 32;    // 4 wave-cols, 32 N each
    v8f acc0 = {0.f, 0.f, 0.f, 0.f, 0.f, 0.f, 0.f, 0.f};
    v8f acc1 = {0.f, 0.f, 0.f, 0.f, 0.f, 0.f, 0.f, 0.f};

    for (int kt = 0; kt < Kc; kt += 32) {
        // ---- phase 1: batched global fetch (branch-free, mask-multiplied) ----
        float av[4];
        #pragma unroll
        for (int t = 0; t < 4; ++t) {
            int id = tid + t * 256;
            int r = id >> 5, c = id & 31;
            int gr = mBase + r, gc = kt + c;
            int grc = gr < M  ? gr : M - 1;
            int gcc = gc < Kc ? gc : Kc - 1;
            float mask = (gr < M && gc < Kc) ? 1.f : 0.f;
            av[t] = A[(long)grc * Kc + gcc] * mask;
        }
        float4 bv[4];
        #pragma unroll
        for (int t = 0; t < 4; ++t) {
            int id = tid + t * 256;
            int r = id >> 5;             // K row 0..31
            int c4 = (id & 31) * 4;      // N col 0..124
            int gr = kt + r;
            int grc = gr < Kc ? gr : Kc - 1;
            float mask = (gr < Kc) ? 1.f : 0.f;
            float4 v = *(const float4*)&Bb[(long)grc * P + nBase + c4];
            bv[t].x = v.x * mask; bv[t].y = v.y * mask;
            bv[t].z = v.z * mask; bv[t].w = v.w * mask;
        }
        // prefetch next K tile of B into cache (global_prefetch_b8)
        if (kt + 32 < Kc) {
            __builtin_prefetch(&Bb[(long)(kt + 32 + (tid >> 3)) * P + nBase + (tid & 7) * 16]);
        }
        // ---- phase 2: convert + LDS stores ----
        #pragma unroll
        for (int t = 0; t < 4; ++t) {
            int id = tid + t * 256;
            int r = id >> 5, c = id & 31;
            int hh = c >> 4;
            int j  = c & 15;
            lAs[r * 40 + hh * 16 + j] = (__bf16)av[t];
        }
        #pragma unroll
        for (int t = 0; t < 4; ++t) {
            int id = tid + t * 256;
            int r = id >> 5;
            int c4 = (id & 31) * 4;
            lBs[(c4 + 0) * 40 + r] = (__bf16)bv[t].x;
            lBs[(c4 + 1) * 40 + r] = (__bf16)bv[t].y;
            lBs[(c4 + 2) * 40 + r] = (__bf16)bv[t].z;
            lBs[(c4 + 3) * 40 + r] = (__bf16)bv[t].w;
        }
        __syncthreads();
        // ---- phase 3: fragments (2x contiguous 16B chunks each) + 2 WMMAs ----
        const __bf16* ap = &lAs[(wm + m) * 40 + h * 16];
        v8bf a0 = *(const v8bf*)ap;
        v8bf a1 = *(const v8bf*)(ap + 8);
        v16bf af = __builtin_shufflevector(a0, a1, 0, 1, 2, 3, 4, 5, 6, 7,
                                                   8, 9, 10, 11, 12, 13, 14, 15);
        const __bf16* bp0 = &lBs[(wn + m) * 40 + h * 16];
        const __bf16* bp1 = &lBs[(wn + 16 + m) * 40 + h * 16];
        v8bf b00 = *(const v8bf*)bp0;
        v8bf b01 = *(const v8bf*)(bp0 + 8);
        v8bf b10 = *(const v8bf*)bp1;
        v8bf b11 = *(const v8bf*)(bp1 + 8);
        v16bf bf0 = __builtin_shufflevector(b00, b01, 0, 1, 2, 3, 4, 5, 6, 7,
                                                      8, 9, 10, 11, 12, 13, 14, 15);
        v16bf bf1 = __builtin_shufflevector(b10, b11, 0, 1, 2, 3, 4, 5, 6, 7,
                                                      8, 9, 10, 11, 12, 13, 14, 15);
        acc0 = __builtin_amdgcn_wmma_f32_16x16x32_bf16(
                   false, af, false, bf0, (short)0, acc0, false, false);
        acc1 = __builtin_amdgcn_wmma_f32_16x16x32_bf16(
                   false, af, false, bf1, (short)0, acc1, false, false);
        __syncthreads();
    }
    float* Cb = Cm + (long)b * strideC;
    int col0 = nBase + wn + m;
    int col1 = col0 + 16;
    #pragma unroll
    for (int r = 0; r < 8; ++r) {
        int row = mBase + wm + r + 8 * h;
        if (row < M) {
            float bv = bias ? bias[row] : 0.f;
            if (col0 < P) {
                float v = act_mode(acc0[r] + bv, mode);
                if (transC) Cb[(long)col0 * M + row] = v;
                else        Cb[(long)row * P + col0] = v;
            }
            if (col1 < P) {
                float v = act_mode(acc1[r] + bv, mode);
                if (transC) Cb[(long)col1 * M + row] = v;
                else        Cb[(long)row * P + col1] = v;
            }
        }
    }
}

// ---------------- vector-neuron BN statistics & apply ----------------
// q layout (B,C,3,P2); stats[c]=sum norm, stats[C+c]=sum norm^2
__global__ __launch_bounds__(256) void k_vn_stats(const float* __restrict__ q,
                                                  int C, long P2, float* stats) {
    __shared__ float s1[256], s2[256];
    int c = blockIdx.y, b = blockIdx.z;
    long p = blockIdx.x * 256L + threadIdx.x;
    float n1 = 0.f, n2 = 0.f;
    if (p < P2) {
        const float* qb = q + (((long)b * C + c) * 3) * P2;
        float q0 = qb[p], q1 = qb[P2 + p], q2 = qb[2 * P2 + p];
        float norm = sqrtf(q0 * q0 + q1 * q1 + q2 * q2) + 1e-6f;
        n1 = norm; n2 = norm * norm;
    }
    s1[threadIdx.x] = n1; s2[threadIdx.x] = n2;
    __syncthreads();
    for (int off = 128; off > 0; off >>= 1) {
        if (threadIdx.x < off) {
            s1[threadIdx.x] += s1[threadIdx.x + off];
            s2[threadIdx.x] += s2[threadIdx.x + off];
        }
        __syncthreads();
    }
    if (threadIdx.x == 0) {
        atomicAdd(&stats[c], s1[0]);
        atomicAdd(&stats[C + c], s2[0]);
    }
}

__global__ void k_vn_apply(const float* __restrict__ q, const float* __restrict__ d,
                           int C, int Cd, long P2, const float* __restrict__ stats,
                           const float* __restrict__ g, const float* __restrict__ bb,
                           float* __restrict__ out) {
    long i = blockIdx.x * (long)blockDim.x + threadIdx.x;
    long total = (long)Bsz * C * P2;
    if (i >= total) return;
    long p = i % P2; long r = i / P2;
    int c = r % C; int b = r / C;
    const float* qb = q + (((long)b * C + c) * 3) * P2;
    int cd = (Cd == 1) ? 0 : c;
    const float* db = d + (((long)b * Cd + cd) * 3) * P2;
    float q0 = qb[p], q1 = qb[P2 + p], q2 = qb[2 * P2 + p];
    float d0 = db[p], d1 = db[P2 + p], d2 = db[2 * P2 + p];
    float norm = sqrtf(q0 * q0 + q1 * q1 + q2 * q2) + 1e-6f;
    float cnt  = (float)((long)Bsz * P2);
    float mean = stats[c] / cnt;
    float var  = stats[C + c] / cnt - mean * mean;
    float nbn  = (norm - mean) * rsqrtf(var + 1e-5f) * g[c] + bb[c];
    float s = nbn / norm;
    float qb0 = q0 * s, qb1 = q1 * s, qb2 = q2 * s;
    float dot = qb0 * d0 + qb1 * d1 + qb2 * d2;
    float dsq = d0 * d0 + d1 * d1 + d2 * d2;
    float coef = dot / (dsq + 1e-6f);
    float o0, o1, o2;
    if (dot >= 0.f) { o0 = qb0; o1 = qb1; o2 = qb2; }
    else { o0 = qb0 - coef * d0; o1 = qb1 - coef * d1; o2 = qb2 - coef * d2; }
    float* ob = out + (((long)b * C + c) * 3) * P2;
    ob[p]          = 0.2f * qb0 + 0.8f * o0;
    ob[P2 + p]     = 0.2f * qb1 + 0.8f * o1;
    ob[2 * P2 + p] = 0.2f * qb2 + 0.8f * o2;
}

// direction max-pool over K: x,d (B,C,3,N,K) -> out view (B,Call,3,N) at channel coff
__global__ void k_pool(const float* __restrict__ x, const float* __restrict__ d,
                       int C, float* __restrict__ out, long obs, int coff) {
    long i = blockIdx.x * (long)blockDim.x + threadIdx.x;
    long total = (long)Bsz * C * Npts;
    if (i >= total) return;
    int n = i % Npts; long r = i / Npts;
    int c = r % C; int b = r / C;
    long NK = (long)Npts * KNN;
    const float* xb = x + (((long)b * C + c) * 3) * NK;
    const float* db = d + (((long)b * C + c) * 3) * NK;
    long base = (long)n * KNN;
    float best = -3.4e38f; int bk = 0;
    for (int k = 0; k < KNN; ++k) {
        long o = base + k;
        float dot = xb[o] * db[o] + xb[NK + o] * db[NK + o] + xb[2 * NK + o] * db[2 * NK + o];
        if (dot > best) { best = dot; bk = k; }
    }
    long oo = base + bk;
    float* ob = out + (long)b * obs + (long)(coff + c) * 3 * Npts;
    ob[n]            = xb[oo];
    ob[Npts + n]     = xb[NK + oo];
    ob[2 * Npts + n] = xb[2 * NK + oo];
}

// mean over N: h (B,C,3,N) -> hm (B,C,3)
__global__ void k_mean_n(const float* __restrict__ h, int C, float* __restrict__ hm) {
    int i = blockIdx.x * blockDim.x + threadIdx.x;
    int total = Bsz * C * 3;
    if (i >= total) return;
    const float* p = h + (long)i * Npts;
    float s = 0.f;
    for (int n = 0; n < Npts; ++n) s += p[n];
    hm[i] = s / (float)Npts;
}

// concat [h (B,341,3,N), mean broadcast] -> (B,682,3,N)
__global__ void k_cat682(const float* __restrict__ h, const float* __restrict__ hm,
                         float* __restrict__ out) {
    long i = blockIdx.x * (long)blockDim.x + threadIdx.x;
    long total = (long)Bsz * 682 * 3 * Npts;
    if (i >= total) return;
    int n = i % Npts; long r = i / Npts;
    int v = r % 3; r /= 3;
    int c = r % 682; int b = r / 682;
    float val;
    if (c < 341) val = h[(((long)b * 341 + c) * 3 + v) * Npts + n];
    else         val = hm[((long)b * 341 + (c - 341)) * 3 + v];
    out[i] = val;
}

// in-place per-point rotation: out[b,i,k,n] = sum_j h[b,i,j,n] * z0pre[b,k,j,n]
__global__ void k_rotate(float* __restrict__ h, int C, const float* __restrict__ z0) {
    long i = blockIdx.x * (long)blockDim.x + threadIdx.x;
    long total = (long)Bsz * C * Npts;
    if (i >= total) return;
    int n = i % Npts; long r = i / Npts;
    int c = r % C; int b = r / C;
    float* hb = h + (((long)b * C + c) * 3) * Npts;
    float h0 = hb[n], h1 = hb[Npts + n], h2 = hb[2 * Npts + n];
    const float* zb = z0 + (long)b * 9 * Npts;
    #pragma unroll
    for (int k = 0; k < 3; ++k) {
        float o = h0 * zb[((long)k * 3 + 0) * Npts + n]
                + h1 * zb[((long)k * 3 + 1) * Npts + n]
                + h2 * zb[((long)k * 3 + 2) * Npts + n];
        hb[(long)k * Npts + n] = o;
    }
}

__global__ void k_rowmax(const float* __restrict__ h, int R, float* __restrict__ out) {
    int i = blockIdx.x * blockDim.x + threadIdx.x;
    if (i >= Bsz * R) return;
    const float* p = h + (long)i * Npts;
    float m = -3.4e38f;
    for (int n = 0; n < Npts; ++n) m = fmaxf(m, p[n]);
    out[i] = m;
}

// build (B,2235,N): rows<2046 broadcast hmax, rest x123rot (B,189,N)
__global__ void k_hcat(const float* __restrict__ hmax, const float* __restrict__ x123,
                       float* __restrict__ out) {
    long i = blockIdx.x * (long)blockDim.x + threadIdx.x;
    long total = (long)Bsz * 2235 * Npts;
    if (i >= total) return;
    int n = i % Npts; long r = i / Npts;
    int c = r % 2235; int b = r / 2235;
    float v;
    if (c < 2046) v = hmax[(long)b * 2046 + c];
    else          v = x123[((long)b * 189 + (c - 2046)) * Npts + n];
    out[i] = v;
}

// per-channel batchnorm stats over (b, n) of h (B,C,N)
__global__ __launch_bounds__(256) void k_bn_stats(const float* __restrict__ h, int C,
                                                  float* stats) {
    __shared__ float s1[256], s2[256];
    int c = blockIdx.y, b = blockIdx.z;
    int n = blockIdx.x * 256 + threadIdx.x;
    float v1 = 0.f, v2 = 0.f;
    if (n < Npts) { float v = h[((long)b * C + c) * Npts + n]; v1 = v; v2 = v * v; }
    s1[threadIdx.x] = v1; s2[threadIdx.x] = v2;
    __syncthreads();
    for (int off = 128; off > 0; off >>= 1) {
        if (threadIdx.x < off) {
            s1[threadIdx.x] += s1[threadIdx.x + off];
            s2[threadIdx.x] += s2[threadIdx.x + off];
        }
        __syncthreads();
    }
    if (threadIdx.x == 0) { atomicAdd(&stats[c], s1[0]); atomicAdd(&stats[C + c], s2[0]); }
}

__global__ void k_bn_apply(float* __restrict__ h, int C, const float* __restrict__ stats,
                           const float* __restrict__ g, const float* __restrict__ bb,
                           float slope) {
    long i = blockIdx.x * (long)blockDim.x + threadIdx.x;
    long total = (long)Bsz * C * Npts;
    if (i >= total) return;
    long r = i / Npts;
    int c = r % C;
    float cnt = (float)(Bsz * Npts);
    float mean = stats[c] / cnt;
    float var  = stats[C + c] / cnt - mean * mean;
    float y = (h[i] - mean) * rsqrtf(var + 1e-5f) * g[c] + bb[c];
    h[i] = y > 0.f ? y : slope * y;
}

// fc (B,256,N) -> out (B,N,256)
__global__ void k_transpose_cf(const float* __restrict__ fc, float* __restrict__ out) {
    long i = blockIdx.x * (long)blockDim.x + threadIdx.x;
    long total = (long)Bsz * Npts * 256;
    if (i >= total) return;
    int m = i % 256; long r = i / 256;
    int n = r % Npts; int b = r / Npts;
    out[i] = fc[((long)b * 256 + m) * Npts + n];
}

// ---------------- host orchestration ----------------
extern "C" void kernel_launch(void* const* d_in, const int* in_sizes, int n_in,
                              void* d_out, int out_size, void* d_ws, size_t ws_size,
                              hipStream_t stream) {
    const float* x = (const float*)d_in[0];
    auto W = [&](int i) { return (const float*)d_in[i]; };

    // workspace layout (floats), 64-float (256B) aligned blocks
    float* ws = (float*)d_ws;
    long o = 0;
    auto alloc = [&](long n) { long r = o; o += (n + 63) & ~63L; return r; };
    float* XT    = ws + alloc((long)Bsz * 3 * Npts);
    float* XX    = ws + alloc((long)Bsz * Npts);
    int*   IDX   = (int*)(ws + alloc((long)Bsz * Npts * KNN));
    float* STATS = ws + alloc(1024);
    float* HM    = ws + alloc((long)Bsz * 341 * 3);
    float* HMAX  = ws + alloc((long)Bsz * 2046);
    float* Z0    = ws + alloc((long)Bsz * 9 * Npts);
    float* X123  = ws + alloc((long)Bsz * 63 * 3 * Npts);
    float* BIGA  = ws + alloc(20643840L);  // graph features / h682
    float* BIGQ  = ws + alloc(18309120L);  // q buffers / hcat / fc
    float* BIGD  = ws + alloc(10321920L);  // d buffers / h8
    float* BIGO  = ws + alloc(10321920L);  // lrelu outputs / h9

    float* out_s = (float*)d_out;
    float* out_c = out_s + (long)Bsz * Npts * 19;
    float* out_v = out_c + (long)Bsz * Npts * 1;
    float* out_f = out_v + (long)Bsz * Npts * 260;

    auto grid1 = [](long total) { return dim3((unsigned)((total + 255) / 256)); };

    auto gemm = [&](const float* A, const float* Bm, float* C, const float* bias,
                    int M, int Kc, int P, long sB, long sC, int mode, int transC) {
        dim3 g((P + 127) / 128, (M + 31) / 32, Bsz);
        k_gemm<<<g, dim3(256), 0, stream>>>(A, Bm, C, bias, M, Kc, P, sB, sC, mode, transC);
    };

    auto vnlayer = [&](int wf, int wd, int gi, int bi, int Cin, int Cout, int Cd,
                       long P2, const float* xin, float* qb, float* db, float* ob) {
        int P = (int)(3 * P2);
        gemm(W(wf), xin, qb, nullptr, Cout, Cin, P, (long)Cin * P, (long)Cout * P, 0, 0);
        gemm(W(wd), xin, db, nullptr, Cd,   Cin, P, (long)Cin * P, (long)Cd   * P, 0, 0);
        k_zero<<<grid1(1024), 256, 0, stream>>>(STATS, 1024);
        k_vn_stats<<<dim3((unsigned)((P2 + 255) / 256), Cout, Bsz), 256, 0, stream>>>(qb, Cout, P2, STATS);
        long total = (long)Bsz * Cout * P2;
        k_vn_apply<<<grid1(total), 256, 0, stream>>>(qb, db, Cout, Cd, P2, STATS, W(gi), W(bi), ob);
    };

    auto knn = [&](const float* X, int D, long bs) {
        k_xx<<<grid1((long)Bsz * Npts), 256, 0, stream>>>(X, D, bs, XX);
        k_knn<<<dim3(Npts, Bsz), 256, 0, stream>>>(X, XX, D, bs, IDX);
    };

    auto graphfeat = [&](const float* F, int C, long bs) {
        long total = (long)Bsz * 2 * C * 3 * Npts * KNN;
        k_graphfeat<<<grid1(total), 256, 0, stream>>>(F, C, bs, IDX, BIGA);
    };

    auto pool = [&](int wdi, const float* xin, int coff) {
        int P = 3 * Npts * KNN;
        gemm(W(wdi), xin, BIGD, nullptr, 21, 21, P, 21L * P, 21L * P, 0, 0);
        long total = (long)Bsz * 21 * Npts;
        k_pool<<<grid1(total), 256, 0, stream>>>(xin, BIGD, 21, X123, 63L * 3 * Npts, coff);
    };

    const long NK = (long)Npts * KNN;
    const long bsX123 = 63L * 3 * Npts;

    // ---- stage 1 ----
    k_transpose_x<<<grid1((long)Bsz * Npts * 3), 256, 0, stream>>>(x, XT);
    knn(XT, 3, 3L * Npts);
    graphfeat(XT, 1, 3L * Npts);                                     // BIGA (B,2,3,N,K)
    vnlayer(1, 2, 3, 4,     2, 21, 21, NK, BIGA, BIGQ, BIGD, BIGO);  // conv1
    vnlayer(5, 6, 7, 8,    21, 21, 21, NK, BIGO, BIGQ, BIGD, BIGA);  // conv2
    pool(34, BIGA, 0);                                               // x1 -> X123[0:21]

    // ---- stage 2 ----
    knn(X123, 63, bsX123);
    graphfeat(X123, 21, bsX123);                                     // BIGA (B,42,3,N,K)
    vnlayer(9, 10, 11, 12, 42, 21, 21, NK, BIGA, BIGQ, BIGD, BIGO);  // conv3
    vnlayer(13, 14, 15, 16, 21, 21, 21, NK, BIGO, BIGQ, BIGD, BIGA); // conv4
    pool(35, BIGA, 21);                                              // x2 -> X123[21:42]

    // ---- stage 3 ----
    knn(X123 + 21L * 3 * Npts, 63, bsX123);
    graphfeat(X123 + 21L * 3 * Npts, 21, bsX123);                    // BIGA (B,42,3,N,K)
    vnlayer(17, 18, 19, 20, 42, 21, 21, NK, BIGA, BIGQ, BIGD, BIGO); // conv5
    pool(36, BIGO, 42);                                              // x3 -> X123[42:63]

    // ---- global vector-neuron stack ----
    vnlayer(21, 22, 23, 24, 63, 341, 1, (long)Npts, X123, BIGQ, BIGD, BIGO);   // conv6 -> h6
    k_mean_n<<<grid1((long)Bsz * 341 * 3), 256, 0, stream>>>(BIGO, 341, HM);
    k_cat682<<<grid1((long)Bsz * 682 * 3 * Npts), 256, 0, stream>>>(BIGO, HM, BIGA); // h682
    vnlayer(25, 26, 27, 28, 682, 341, 341, (long)Npts, BIGA, BIGQ, BIGD, BIGO); // vn1 -> z1
    vnlayer(29, 30, 31, 32, 341, 170, 170, (long)Npts, BIGO, BIGQ, BIGD, BIGO); // vn2 -> z2
    gemm(W(33), BIGO, Z0, nullptr, 3, 170, 3 * Npts, 170L * 3 * Npts, 9L * Npts, 0, 0); // z0

    // rotate h682 and x123 in place; then global max + concat
    k_rotate<<<grid1((long)Bsz * 682 * Npts), 256, 0, stream>>>(BIGA, 682, Z0);
    k_rotate<<<grid1((long)Bsz * 63 * Npts), 256, 0, stream>>>(X123, 63, Z0);
    k_rowmax<<<grid1((long)Bsz * 2046), 256, 0, stream>>>(BIGA, 2046, HMAX);
    k_hcat<<<grid1((long)Bsz * 2235 * Npts), 256, 0, stream>>>(HMAX, X123, BIGQ); // (B,2235,N)

    // ---- dense head ----
    gemm(W(37), BIGQ, BIGD, nullptr, 256, 2235, Npts, 2235L * Npts, 256L * Npts, 0, 0); // conv8
    k_zero<<<grid1(1024), 256, 0, stream>>>(STATS, 1024);
    k_bn_stats<<<dim3(Npts / 256, 256, Bsz), 256, 0, stream>>>(BIGD, 256, STATS);
    k_bn_apply<<<grid1((long)Bsz * 256 * Npts), 256, 0, stream>>>(BIGD, 256, STATS, W(38), W(39), 0.2f);

    gemm(W(40), BIGD, BIGO, nullptr, 256, 256, Npts, 256L * Npts, 256L * Npts, 0, 0);   // conv9
    k_zero<<<grid1(1024), 256, 0, stream>>>(STATS, 1024);
    k_bn_stats<<<dim3(Npts / 256, 256, Bsz), 256, 0, stream>>>(BIGO, 256, STATS);
    k_bn_apply<<<grid1((long)Bsz * 256 * Npts), 256, 0, stream>>>(BIGO, 256, STATS, W(41), W(42), 0.2f);

    gemm(W(43), BIGO, BIGQ, W(44), 256, 256, Npts, 256L * Npts, 256L * Npts, 2, 0);     // mlp -> fc
    k_transpose_cf<<<grid1((long)Bsz * Npts * 256), 256, 0, stream>>>(BIGQ, out_f);     // f

    gemm(W(47), BIGQ, out_s, W(48), 19, 256, Npts, 256L * Npts, (long)Npts * 19, 2, 1);  // s
    gemm(W(49), BIGQ, out_c, W(50),  1, 256, Npts, 256L * Npts, (long)Npts,      3, 1);  // c
    gemm(W(45), BIGQ, out_v, W(46), 260, 256, Npts, 256L * Npts, (long)Npts * 260, 4, 1);// v
}
